// ResnetSpherePointHeightnet_63788854280418
// MI455X (gfx1250) — matrix operand order repaired and checked by reading, not compiled, not run on hardware
//
#include <hip/hip_runtime.h>
#include <hip/hip_bf16.h>

// ---------------------------------------------------------------------------
// MI455X (gfx1250) implementation.
//  - PointNet resblocks: fused bf16 WMMA kernel (v_wmma_f32_16x16x32_bf16),
//    wave32, 8 waves/WG, each wave computes a 16x128 output tile in-place.
//    fc0/shortcut weights staged once per WG into LDS (160KB/WG -> 2 WGs per
//    320KB WGP): cuts per-resblock L2 weight traffic 8x (3.3GB -> 0.4GB) and
//    turns the B-fragment fetches into short-latency ds_load_b128.
//    Packed relu via inline v_pk_max_num_bf16.
//  - Scatter: monotonic-uint-encoded float atomic max/min + float atomicAdd.
//  - UNet: direct conv kernels (L2-resident, <4% of FLOPs).
// ---------------------------------------------------------------------------

typedef unsigned int  u32;
typedef unsigned short u16;

typedef __attribute__((ext_vector_type(16))) __bf16 v16bf;
typedef __attribute__((ext_vector_type(8)))  float  v8f;

struct alignas(16) U128 { u32 u[4]; };

union Frag {           // 32 bytes: one 16-element bf16 WMMA operand
    U128  q[2];
    u32   u[8];
    v16bf v;
};

#define M_TOTAL 400000
#define T_PTS   100000

// ------------------------- small helpers -----------------------------------

__device__ __forceinline__ u16 f2bf(float f) {          // round-to-nearest-even
    u32 u = __float_as_uint(f);
    u += 0x7FFFu + ((u >> 16) & 1u);
    return (u16)(u >> 16);
}
__device__ __forceinline__ float bf2f(u16 h) { return __uint_as_float(((u32)h) << 16); }

// order-preserving float <-> uint encoding for atomic max/min
__device__ __forceinline__ u32 fenc(float f) {
    u32 u = __float_as_uint(f);
    return (u & 0x80000000u) ? ~u : (u | 0x80000000u);
}
__device__ __forceinline__ float fdec(u32 e) {
    u32 u = (e & 0x80000000u) ? (e & 0x7FFFFFFFu) : ~e;
    return __uint_as_float(u);
}

// relu on a packed pair of bf16: single CDNA5 VOP3P instruction.
// (V_PK_MAX_NUM_BF16, ISA 15.10 opcode 44; inline constant 0 == packed +0.0)
__device__ __forceinline__ u32 relu_pk(u32 w) {
    u32 r;
    asm("v_pk_max_num_bf16 %0, %1, 0" : "=v"(r) : "v"(w));
    return r;
}

// 16-byte chunk of a "virtual" 256-wide bf16 row split as [lo(128) | hi(128)]
__device__ __forceinline__ U128 ldx16(const u16* __restrict__ lo,
                                      const u16* __restrict__ hi,
                                      int bh, int batch, int row, int k) {
    if (k < 128) return *(const U128*)(lo + (size_t)row * 128 + k);
    const u16* base = bh ? (hi + (size_t)batch * 128) : (hi + (size_t)row * 128);
    return *(const U128*)(base + (k - 128));
}

// --------------------- weight pre-transform (f32 -> bf16, transposed) ------

__global__ void prep_weights(const float* __restrict__ w0,
                             const float* __restrict__ w1,
                             const float* __restrict__ ws,
                             u16* __restrict__ w0t, u16* __restrict__ w1t,
                             u16* __restrict__ wst) {
    int idx = blockIdx.x * blockDim.x + threadIdx.x;
    if (idx < 163840) {                        // blk_fc0_w (5,256,128) -> [blk][n][k]
        int blk = idx / 32768, rem = idx % 32768, k = rem / 128, n = rem % 128;
        w0t[blk * 32768 + n * 256 + k] = f2bf(w0[idx]);
    } else if (idx < 327680) {                 // blk_sc_w (5,256,128)
        int j = idx - 163840;
        int blk = j / 32768, rem = j % 32768, k = rem / 128, n = rem % 128;
        wst[blk * 32768 + n * 256 + k] = f2bf(ws[j]);
    } else if (idx < 409600) {                 // blk_fc1_w (5,128,128)
        int j = idx - 327680;
        int blk = j / 16384, rem = j % 16384, k = rem / 128, n = rem % 128;
        w1t[blk * 16384 + n * 128 + k] = f2bf(w1[j]);
    }
}

// --------------------------- fc_pos ----------------------------------------

__global__ void fc_pos_kernel(const float* __restrict__ p,
                              const float* __restrict__ w,   // (3,256)
                              const float* __restrict__ b,
                              u16* __restrict__ lo, u16* __restrict__ hi) {
    int row = blockIdx.x * blockDim.x + threadIdx.x;
    if (row >= M_TOTAL) return;
    float x = p[row * 3 + 0], y = p[row * 3 + 1], z = p[row * 3 + 2];
    for (int c = 0; c < 256; ++c) {
        float v = x * w[c] + y * w[256 + c] + z * w[512 + c] + b[c];
        if (c < 128) lo[(size_t)row * 128 + c] = f2bf(v);
        else         hi[(size_t)row * 128 + (c - 128)] = f2bf(v);
    }
}

// --------------------- fused ResnetBlockFC via WMMA -------------------------
// out = x @ Ws + relu(relu(x)@W0 + b0) @ W1 + b1     (written in-place capable)

__global__ void __launch_bounds__(256)
resblock_wmma(const u16* __restrict__ xlo, const u16* __restrict__ xhi, int bh,
              const u16* __restrict__ w0t, const float* __restrict__ b0,
              const u16* __restrict__ w1t, const float* __restrict__ b1,
              const u16* __restrict__ wst,
              u16* __restrict__ out) {
    // 160KB LDS total -> 2 workgroups per 320KB WGP
    __shared__ alignas(16) u16 s_w0t[128 * 256];      // 64KB, [n][k] bf16
    __shared__ alignas(16) u16 s_wst[128 * 256];      // 64KB, [n][k] bf16
    __shared__ alignas(16) u16 lds_net[8][16][128];   // 32KB, per-wave net tile

    // ---- cooperative weight staging: 128KB from L2, once per workgroup ----
    {
        const U128* g0 = (const U128*)w0t;
        const U128* gs = (const U128*)wst;
        U128* l0 = (U128*)s_w0t;
        U128* ls = (U128*)s_wst;
#pragma unroll 4
        for (int i = threadIdx.x; i < 4096; i += 256) l0[i] = g0[i];
#pragma unroll 4
        for (int i = threadIdx.x; i < 4096; i += 256) ls[i] = gs[i];
    }
    __syncthreads();

    const int lane  = threadIdx.x & 31;
    const int wave  = threadIdx.x >> 5;
    const int row0  = blockIdx.x * 128 + wave * 16;   // M_TOTAL % 128 == 0
    const int batch = row0 / T_PTS;                   // tile never crosses a batch
    const int ncol  = lane & 15;
    const int hi16  = (lane & 16) ? 1 : 0;
    const int mrow  = row0 + ncol;                    // this lane's A row

    __builtin_prefetch(xlo + (size_t)mrow * 128, 0, 0);

    // ---------------- phase A: acc = relu(x) @ W0 (K = 256) ----------------
    v8f acc[8];
#pragma unroll
    for (int nt = 0; nt < 8; ++nt) acc[nt] = (v8f){0.f,0.f,0.f,0.f,0.f,0.f,0.f,0.f};

    for (int kt = 0; kt < 8; ++kt) {
        Frag a;
        int k1 = kt * 32 + (hi16 ? 8 : 0);
        a.q[0] = ldx16(xlo, xhi, bh, batch, mrow, k1);
        a.q[1] = ldx16(xlo, xhi, bh, batch, mrow, k1 + 16);
#pragma unroll
        for (int i = 0; i < 8; ++i) a.u[i] = relu_pk(a.u[i]);
        int kb = kt * 32 + (hi16 ? 8 : 0);
#pragma unroll
        for (int nt = 0; nt < 8; ++nt) {
            Frag bfr;
            const U128* wp = (const U128*)(s_w0t + (nt * 16 + ncol) * 256 + kb);
            bfr.q[0] = wp[0]; bfr.q[1] = wp[2];
            acc[nt] = __builtin_amdgcn_wmma_f32_16x16x32_bf16(
                false, a.v, false, bfr.v, (short)0, acc[nt], false, false);
        }
    }

    // net = relu(acc + b0) -> per-wave LDS tile (bf16), feeds fc1 A-fragments
#pragma unroll
    for (int nt = 0; nt < 8; ++nt) {
        float bias = b0[nt * 16 + ncol];
#pragma unroll
        for (int r = 0; r < 8; ++r) {
            float v = acc[nt][r] + bias;
            v = v > 0.f ? v : 0.f;
            lds_net[wave][hi16 * 8 + r][nt * 16 + ncol] = f2bf(v);
        }
    }

    // -------- phase B: acc = b1 + x @ Ws (K=256) + net @ W1 (K=128) --------
#pragma unroll
    for (int nt = 0; nt < 8; ++nt) {
        float bias = b1[nt * 16 + ncol];
        v8f c;
#pragma unroll
        for (int r = 0; r < 8; ++r) c[r] = bias;
        acc[nt] = c;
    }

    for (int kt = 0; kt < 8; ++kt) {                   // shortcut: raw x
        Frag a;
        int k1 = kt * 32 + (hi16 ? 8 : 0);
        a.q[0] = ldx16(xlo, xhi, bh, batch, mrow, k1);
        a.q[1] = ldx16(xlo, xhi, bh, batch, mrow, k1 + 16);
        int kb = kt * 32 + (hi16 ? 8 : 0);
#pragma unroll
        for (int nt = 0; nt < 8; ++nt) {
            Frag bfr;
            const U128* wp = (const U128*)(s_wst + (nt * 16 + ncol) * 256 + kb);
            bfr.q[0] = wp[0]; bfr.q[1] = wp[2];
            acc[nt] = __builtin_amdgcn_wmma_f32_16x16x32_bf16(
                false, a.v, false, bfr.v, (short)0, acc[nt], false, false);
        }
    }

    for (int kt = 0; kt < 4; ++kt) {                   // fc1: A from LDS tile
        Frag a;
        int k1 = kt * 32 + (hi16 ? 8 : 0);
        a.q[0] = *(const U128*)&lds_net[wave][ncol][k1];
        a.q[1] = *(const U128*)&lds_net[wave][ncol][k1 + 16];
        int kb = kt * 32 + (hi16 ? 8 : 0);
#pragma unroll
        for (int nt = 0; nt < 8; ++nt) {
            Frag bfr;                                   // fc1 B stays in L2
            const U128* wp = (const U128*)(w1t + (size_t)(nt * 16 + ncol) * 128 + kb);
            bfr.q[0] = wp[0]; bfr.q[1] = wp[2];
            acc[nt] = __builtin_amdgcn_wmma_f32_16x16x32_bf16(
                false, a.v, false, bfr.v, (short)0, acc[nt], false, false);
        }
    }

    // ---------------- write out (stage through LDS, vector stores) ---------
#pragma unroll
    for (int nt = 0; nt < 8; ++nt)
#pragma unroll
        for (int r = 0; r < 8; ++r)
            lds_net[wave][hi16 * 8 + r][nt * 16 + ncol] = f2bf(acc[nt][r]);

    {
        const U128* src = (const U128*)&lds_net[wave][ncol][hi16 * 64];
        U128* dst = (U128*)(out + (size_t)(row0 + ncol) * 128 + hi16 * 64);
        dst[0] = src[0]; dst[1] = src[1]; dst[2] = src[2]; dst[3] = src[3];
    }
}

// --------------------- per-batch column max (pooled) ------------------------

__global__ void pooled_init(u32* penc) {
    int i = blockIdx.x * blockDim.x + threadIdx.x;
    if (i < 512) penc[i] = fenc(-__builtin_inff());
}

__global__ void __launch_bounds__(128)
rowmax_kernel(const u16* __restrict__ net, u32* __restrict__ penc) {
    int c = threadIdx.x;                    // channel 0..127
    int r0 = blockIdx.x * 256;
    int cur_b = -1;
    float m = -__builtin_inff();
    for (int i = 0; i < 256; ++i) {
        int row = r0 + i;
        if (row >= M_TOTAL) break;
        int b = row / T_PTS;
        if (b != cur_b) {
            if (cur_b >= 0) atomicMax(&penc[cur_b * 128 + c], fenc(m));
            cur_b = b; m = -__builtin_inff();
        }
        float v = bf2f(net[(size_t)row * 128 + c]);
        m = v > m ? v : m;
    }
    if (cur_b >= 0) atomicMax(&penc[cur_b * 128 + c], fenc(m));
}

__global__ void pooled_cvt(const u32* penc, u16* pbf) {
    int i = blockIdx.x * blockDim.x + threadIdx.x;
    if (i < 512) pbf[i] = f2bf(fdec(penc[i]));
}

// ----------------------------- scatter --------------------------------------

__global__ void scatter_init(u32* fmaxe, u32* hmaxe, u32* hmine,
                             float* hsum, float* hcnt) {
    int i = blockIdx.x * blockDim.x + threadIdx.x;
    if (i < 2097152) fmaxe[i] = 0x80000000u;           // fenc(0.0f)
    if (i < 16384) {
        hmaxe[i] = 0x80000000u;
        hmine[i] = 0x80000000u;
        hsum[i] = 0.f; hcnt[i] = 0.f;
    }
}

__global__ void scatter_kernel(const float* __restrict__ p,
                               const u16* __restrict__ net,
                               u32* fmaxe, u32* hmaxe, u32* hmine,
                               float* hsum, float* hcnt) {
    int i = blockIdx.x * blockDim.x + threadIdx.x;
    if (i >= M_TOTAL) return;
    int b = i / T_PTS;
    float x = p[i * 3 + 0], y = p[i * 3 + 1], z = p[i * 3 + 2];
    float r = sqrtf(x * x + y * y + z * z);
    const float PI = 3.1415927410125732f;
    float lat = 90.f - atan2f(z, sqrtf(x * x + y * y)) * (180.f / PI);
    float mer = fmodf(360.f + atan2f(y, x) * (180.f / PI), 360.f);
    int yg = (int)floorf(lat * (64.f / 180.f));
    int xg = (int)floorf(mer * (64.f / 360.f));
    int cell = xg + 64 * yg;
    cell = cell < 0 ? 0 : (cell > 4095 ? 4095 : cell);
    int gc = b * 4096 + cell;
    atomicMax(&hmaxe[gc], fenc(r));
    atomicMin(&hmine[gc], fenc(r));
    atomicAdd(&hsum[gc], r);
    atomicAdd(&hcnt[gc], 1.0f);
    u32* fb = fmaxe + (size_t)gc * 128;
    const u16* nr = net + (size_t)i * 128;
    for (int c = 0; c < 128; ++c) atomicMax(&fb[c], fenc(bf2f(nr[c])));
}

__global__ void build_unet_input(const u32* fmaxe, const u32* hmaxe,
                                 const u32* hmine, const float* hsum,
                                 const float* hcnt, float* __restrict__ X) {
    int i = blockIdx.x * blockDim.x + threadIdx.x;
    if (i >= 16 * 131 * 4096) return;
    int s = i % 4096;
    int ch = (i / 4096) % 131;
    int bn = i / (4096 * 131);
    float v = 0.f;
    if ((bn & 3) == 0) {                    // only NCH slot 0 is populated
        int b = bn >> 2;
        int g = b * 4096 + s;
        if (ch < 128)       v = fdec(fmaxe[(size_t)g * 128 + ch]);
        else if (ch == 128) v = fdec(hmine[g]);
        else if (ch == 129) { float c = hcnt[g]; v = hsum[g] / (c > 1.f ? c : 1.f); }
        else                v = fdec(hmaxe[g]);
    }
    X[i] = v;
}

// ------------------------------- UNet ---------------------------------------

__global__ void conv3x3_kernel(const float* __restrict__ in0, int c0,
                               const float* __restrict__ in1, int c1,
                               const float* __restrict__ w,
                               const float* __restrict__ bias,
                               float* __restrict__ out, int S, int Cout) {
    int idx = blockIdx.x * blockDim.x + threadIdx.x;
    int total = 16 * Cout * S * S;
    if (idx >= total) return;
    int x = idx % S, y = (idx / S) % S;
    int co = (idx / (S * S)) % Cout, n = idx / (S * S * Cout);
    int Cin = c0 + c1;
    float acc = bias[co];
    for (int ky = -1; ky <= 1; ++ky) {
        int iy = y + ky; if (iy < 0 || iy >= S) continue;
        for (int kx = -1; kx <= 1; ++kx) {
            int ix = x + kx; if (ix < 0 || ix >= S) continue;
            const float* wp = w + (size_t)co * Cin * 9 + (ky + 1) * 3 + (kx + 1);
            for (int ci = 0; ci < c0; ++ci)
                acc += in0[(((size_t)n * c0 + ci) * S + iy) * S + ix] * wp[(size_t)ci * 9];
            for (int ci = 0; ci < c1; ++ci)
                acc += in1[(((size_t)n * c1 + ci) * S + iy) * S + ix] * wp[(size_t)(c0 + ci) * 9];
        }
    }
    out[idx] = acc > 0.f ? acc : 0.f;       // every 3x3 conv in the net is relu'd
}

__global__ void pool2_kernel(const float* __restrict__ in, float* __restrict__ out,
                             int C, int So) {
    int idx = blockIdx.x * blockDim.x + threadIdx.x;
    int total = 16 * C * So * So;
    if (idx >= total) return;
    int x = idx % So, y = (idx / So) % So;
    int c = (idx / (So * So)) % C, n = idx / (So * So * C);
    int Si = So * 2;
    const float* ip = in + (((size_t)n * C + c) * Si + 2 * y) * Si + 2 * x;
    float m = ip[0];
    m = fmaxf(m, ip[1]); m = fmaxf(m, ip[Si]); m = fmaxf(m, ip[Si + 1]);
    out[idx] = m;
}

// 2x2 stride-2 transpose conv: each output taps exactly one input pixel
__global__ void deconv2_kernel(const float* __restrict__ in, int Cin,
                               const float* __restrict__ w,
                               const float* __restrict__ bias,
                               float* __restrict__ out, int So, int Cout) {
    int idx = blockIdx.x * blockDim.x + threadIdx.x;
    int total = 16 * Cout * So * So;
    if (idx >= total) return;
    int x = idx % So, y = (idx / So) % So;
    int co = (idx / (So * So)) % Cout, n = idx / (So * So * Cout);
    int Si = So >> 1;
    int iy = y >> 1, ix = x >> 1, ky = y & 1, kx = x & 1;
    float acc = bias[co];
    for (int ci = 0; ci < Cin; ++ci)
        acc += in[(((size_t)n * Cin + ci) * Si + iy) * Si + ix] *
               w[(((size_t)co * Cin + ci) * 2 + ky) * 2 + kx];
    out[idx] = acc;
}

// 1x1 out conv, fused with the final NCHW -> [B,NCH,H,W,CDIM] transpose
__global__ void conv1x1_out_kernel(const float* __restrict__ in,
                                   const float* __restrict__ w,
                                   const float* __restrict__ bias,
                                   float* __restrict__ dout) {
    int idx = blockIdx.x * blockDim.x + threadIdx.x;
    if (idx >= 8388608) return;
    int co = idx & 127;
    int x = (idx >> 7) & 63;
    int y = (idx >> 13) & 63;
    int n = idx >> 19;                       // bn 0..15
    int sp = y * 64 + x;
    float acc = bias[co];
    for (int ci = 0; ci < 32; ++ci)
        acc += in[(((size_t)n * 32 + ci) << 12) + sp] * w[co * 32 + ci];
    dout[idx] = acc;
    if (idx < 2) dout[8388608 + idx] = 0.f;  // C_mat = zeros((1,2))
}

// ------------------------------- launcher ------------------------------------

extern "C" void kernel_launch(void* const* d_in, const int* in_sizes, int n_in,
                              void* d_out, int out_size, void* d_ws, size_t ws_size,
                              hipStream_t stream) {
    (void)in_sizes; (void)n_in; (void)out_size; (void)ws_size;

    const float* p        = (const float*)d_in[0];
    const float* fc_pos_w = (const float*)d_in[1];
    const float* fc_pos_b = (const float*)d_in[2];
    const float* fc0_w    = (const float*)d_in[3];
    const float* fc0_b    = (const float*)d_in[4];
    const float* fc1_w    = (const float*)d_in[5];
    const float* fc1_b    = (const float*)d_in[6];
    const float* sc_w     = (const float*)d_in[7];
    const float* we1 = (const float*)d_in[8],  *be1 = (const float*)d_in[9];
    const float* we2 = (const float*)d_in[10], *be2 = (const float*)d_in[11];
    const float* wb  = (const float*)d_in[12], *bb  = (const float*)d_in[13];
    const float* wu2 = (const float*)d_in[14], *bu2 = (const float*)d_in[15];
    const float* wd2 = (const float*)d_in[16], *bd2 = (const float*)d_in[17];
    const float* wu1 = (const float*)d_in[18], *bu1 = (const float*)d_in[19];
    const float* wd1 = (const float*)d_in[20], *bd1 = (const float*)d_in[21];
    const float* wout = (const float*)d_in[22], *bout = (const float*)d_in[23];

    char* ws = (char*)d_ws;
    const size_t NB = (size_t)M_TOTAL * 128 * 2;       // 102,400,000 B

    // BUF0: activation chain (in-place after block 0); final net lives here
    u16* BUF0 = (u16*)ws;
    // BUF1: fc_pos hi half; after block 0 it is dead -> scatter grids reuse it
    u16* BUF1 = (u16*)(ws + NB);
    u32*   HMAXE = (u32*)(ws + NB);
    u32*   HMINE = (u32*)(ws + NB + 65536);
    float* HSUM  = (float*)(ws + NB + 131072);
    float* HCNT  = (float*)(ws + NB + 196608);
    u32*   FMAXE = (u32*)(ws + NB + 262144);           // 8,388,608 B

    // UNet region (~77.4 MB)
    float* X  = (float*)(ws + 2 * NB);
    float* E1 = X  + 8585216;
    float* P1 = E1 + 2097152;
    float* E2 = P1 + 524288;
    float* P2 = E2 + 1048576;
    float* BO = P2 + 262144;
    float* U2 = BO + 524288;
    float* D2 = U2 + 1048576;
    float* U1 = D2 + 1048576;
    float* D1 = U1 + 2097152;

    char* wtail = (char*)(D1 + 2097152);
    u16* W0T = (u16*)wtail;  wtail += 5 * 128 * 256 * 2;
    u16* WST = (u16*)wtail;  wtail += 5 * 128 * 256 * 2;
    u16* W1T = (u16*)wtail;  wtail += 5 * 128 * 128 * 2;
    u32* PENC = (u32*)wtail; wtail += 2048;
    u16* PBF  = (u16*)wtail;

    const int TB = 256;
    auto cdiv = [](long n, int d) { return (int)((n + d - 1) / d); };

    prep_weights<<<cdiv(409600, TB), TB, 0, stream>>>(fc0_w, fc1_w, sc_w, W0T, W1T, WST);
    fc_pos_kernel<<<cdiv(M_TOTAL, TB), TB, 0, stream>>>(p, fc_pos_w, fc_pos_b, BUF0, BUF1);

    // block 0: x = [BUF0 | BUF1] per-row -> BUF0 (in-place lo is safe: each
    // wave reads only its own 16 rows before the final write)
    resblock_wmma<<<M_TOTAL / 128, TB, 0, stream>>>(
        BUF0, BUF1, 0, W0T, fc0_b, W1T, fc1_b, WST, BUF0);

    for (int blk = 1; blk < 5; ++blk) {
        pooled_init<<<2, TB, 0, stream>>>(PENC);
        rowmax_kernel<<<cdiv(M_TOTAL, 256), 128, 0, stream>>>(BUF0, PENC);
        pooled_cvt<<<2, TB, 0, stream>>>(PENC, PBF);
        resblock_wmma<<<M_TOTAL / 128, TB, 0, stream>>>(
            BUF0, PBF, 1,
            W0T + blk * 32768, fc0_b + blk * 128,
            W1T + blk * 16384, fc1_b + blk * 128,
            WST + blk * 32768, BUF0);
    }

    scatter_init<<<cdiv(2097152, TB), TB, 0, stream>>>(FMAXE, HMAXE, HMINE, HSUM, HCNT);
    scatter_kernel<<<cdiv(M_TOTAL, TB), TB, 0, stream>>>(p, BUF0, FMAXE, HMAXE, HMINE, HSUM, HCNT);
    build_unet_input<<<cdiv((long)16 * 131 * 4096, TB), TB, 0, stream>>>(
        FMAXE, HMAXE, HMINE, HSUM, HCNT, X);

    conv3x3_kernel<<<cdiv((long)16 * 32 * 64 * 64, TB), TB, 0, stream>>>(
        X, 131, nullptr, 0, we1, be1, E1, 64, 32);
    pool2_kernel<<<cdiv((long)16 * 32 * 32 * 32, TB), TB, 0, stream>>>(E1, P1, 32, 32);
    conv3x3_kernel<<<cdiv((long)16 * 64 * 32 * 32, TB), TB, 0, stream>>>(
        P1, 32, nullptr, 0, we2, be2, E2, 32, 64);
    pool2_kernel<<<cdiv((long)16 * 64 * 16 * 16, TB), TB, 0, stream>>>(E2, P2, 64, 16);
    conv3x3_kernel<<<cdiv((long)16 * 128 * 16 * 16, TB), TB, 0, stream>>>(
        P2, 64, nullptr, 0, wb, bb, BO, 16, 128);
    deconv2_kernel<<<cdiv((long)16 * 64 * 32 * 32, TB), TB, 0, stream>>>(
        BO, 128, wu2, bu2, U2, 32, 64);
    conv3x3_kernel<<<cdiv((long)16 * 64 * 32 * 32, TB), TB, 0, stream>>>(
        U2, 64, E2, 64, wd2, bd2, D2, 32, 64);
    deconv2_kernel<<<cdiv((long)16 * 32 * 64 * 64, TB), TB, 0, stream>>>(
        D2, 64, wu1, bu1, U1, 64, 32);
    conv3x3_kernel<<<cdiv((long)16 * 32 * 64 * 64, TB), TB, 0, stream>>>(
        U1, 32, E1, 32, wd1, bd1, D1, 64, 32);
    conv1x1_out_kernel<<<cdiv(8388608, TB), TB, 0, stream>>>(D1, wout, bout, (float*)d_out);
}
